// GATEncoder_2284922601880
// MI455X (gfx1250) — matrix operand order repaired and checked by reading, not compiled, not run on hardware
//
#include <hip/hip_runtime.h>
#include <hip/hip_bf16.h>
#include <math.h>

typedef __attribute__((ext_vector_type(16))) _Float16 v16h;
typedef __attribute__((ext_vector_type(8)))  _Float16 v8h;
typedef __attribute__((ext_vector_type(8)))  float    v8f;

#define SLOPE 0.2f

// ---------------------------------------------------------------------------
// Dense GEMM: Y[M,Cout] = X[M,K] @ W[K,Cout] + bias, f16 WMMA with f32 accum.
// Block: 64 rows of X; (Cout/16) waves, wave w owns output cols [16w,16w+16)
// across 4 row tiles. X tile row-major, W transposed ([col][k]) in LDS, both
// padded to KP=K+8 halves so every fragment load is a 16B-aligned b128 with
// conflict-free bank striding.
// ---------------------------------------------------------------------------
template <int K, int COUT>
__global__ __launch_bounds__((COUT / 16) * 32) void gemm_bias_wmma(
    const float* __restrict__ X, const float* __restrict__ W,
    const float* __restrict__ bias, float* __restrict__ Y, int M) {
  constexpr int ROWS = 64;
  constexpr int KP = K + 8;  // padded stride in halves; KP*2 bytes is 16B-aligned
  __shared__ __align__(16) _Float16 sA[ROWS * KP];  // [row][k]
  __shared__ __align__(16) _Float16 sW[COUT * KP];  // [col][k] (transposed)

  const int row0 = blockIdx.x * ROWS;
  const int tid = threadIdx.x;
  const int nthr = blockDim.x;

  // Stage X tile (f32 -> f16), row-major with padded stride
  for (int i = tid; i < ROWS * K; i += nthr) {
    int r = i / K, k = i - r * K;
    int gr = row0 + r;
    float v = (gr < M) ? X[(long)gr * K + k] : 0.0f;
    sA[r * KP + k] = (_Float16)v;
  }
  // Stage W transposed (f32 -> f16): W is [K][COUT] row-major in global
  for (int i = tid; i < K * COUT; i += nthr) {
    int k = i / COUT, c = i - k * COUT;
    sW[c * KP + k] = (_Float16)W[i];
  }
  __syncthreads();

  const int wave = tid >> 5;
  const int lane = tid & 31;
  const int col0 = wave * 16;
  const int half = lane >> 4;   // 0: lanes 0-15, 1: lanes 16-31
  const int mrow = lane & 15;   // A row within tile this lane holds
  const int ncol = lane & 15;   // B/C/D column this lane holds
  const int akb = half * 8;     // A k-base within 32-chunk (ISA 16-bit A layout)
  const int bkb = half * 8;     // B k-base within 32-chunk (matches A layout)

  v8f acc[4] = {};

  for (int k0 = 0; k0 < K; k0 += 32) {
    // B fragment: halves [k0+bkb, +8) and [k0+bkb+16, +8) of sW[(col0+ncol)*KP]
    union { v16h v; v8h h[2]; } ub;
    const v8h* pb = (const v8h*)&sW[(col0 + ncol) * KP + k0 + bkb];
    ub.h[0] = pb[0];
    ub.h[1] = pb[2];
#pragma unroll
    for (int rt = 0; rt < 4; ++rt) {
      // A fragment: halves [k0+akb, +8) and [k0+akb+16, +8)
      union { v16h v; v8h h[2]; } ua;
      const _Float16* base = &sA[(rt * 16 + mrow) * KP + k0 + akb];
      ua.h[0] = *(const v8h*)(base);
      ua.h[1] = *(const v8h*)(base + 16);
      acc[rt] = __builtin_amdgcn_wmma_f32_16x16x32_f16(
          /*neg_a=*/false, ua.v, /*neg_b=*/false, ub.v,
          /*c_mod=*/(short)0, acc[rt], /*reuse_a=*/false, /*reuse_b=*/false);
    }
  }

  // Store D: VGPR r holds row (rt*16 + r + 8*half), col = ncol
  const int gn = col0 + ncol;
  const float bv = bias[gn];
#pragma unroll
  for (int rt = 0; rt < 4; ++rt) {
#pragma unroll
    for (int r = 0; r < 8; ++r) {
      int gm = row0 + rt * 16 + r + half * 8;
      if (gm < M) Y[(long)gm * COUT + gn] = acc[rt][r] + bv;
    }
  }
}

// ---------------------------------------------------------------------------
// Helpers / edge-phase kernels
// ---------------------------------------------------------------------------
__global__ void fill_f32(float* __restrict__ p, long n, float v) {
  long i = (long)blockIdx.x * blockDim.x + threadIdx.x;
  long stride = (long)gridDim.x * blockDim.x;
  for (; i < n; i += stride) p[i] = v;
}

__device__ __forceinline__ void atomicMaxF32(float* addr, float val) {
  // Sign-aware int trick; valid with -inf init.
  if (!signbit(val)) {
    atomicMax((int*)addr, __float_as_int(val));
  } else {
    atomicMin((unsigned int*)addr, __float_as_uint(val));
  }
}

__device__ __forceinline__ void edge_ids(const int* __restrict__ ei, int e, int E,
                                         int& src, int& dst) {
  if (e < E) {
    src = ei[e];
    dst = ei[E + e];
  } else {
    src = dst = e - E;  // self loop
  }
}

// Pass 1: e = dot(leaky_relu(xl[src]+xr[dst]), att); segment max into EMAX.
__global__ void edge_logits(const int* __restrict__ ei, int E, int T,
                            const float* __restrict__ XL, const float* __restrict__ XR,
                            const float* __restrict__ att, int C,
                            float* __restrict__ Ebuf, float* __restrict__ EMAX) {
  int e = blockIdx.x * blockDim.x + threadIdx.x;
  if (e >= T) return;
  int src, dst;
  edge_ids(ei, e, E, src, dst);
  const float4* pl = (const float4*)(XL + (long)src * C);
  const float4* pr = (const float4*)(XR + (long)dst * C);
  const float4* pa = (const float4*)att;
  float s = 0.0f;
  for (int q = 0; q < C / 4; ++q) {
    float4 l = pl[q], r = pr[q], a = pa[q];
    float m;
    m = l.x + r.x; m = m > 0.0f ? m : SLOPE * m; s += m * a.x;
    m = l.y + r.y; m = m > 0.0f ? m : SLOPE * m; s += m * a.y;
    m = l.z + r.z; m = m > 0.0f ? m : SLOPE * m; s += m * a.z;
    m = l.w + r.w; m = m > 0.0f ? m : SLOPE * m; s += m * a.w;
  }
  Ebuf[e] = s;
  atomicMaxF32(&EMAX[dst], s);
}

// Pass 2: ee = exp(e - emax[dst]); segment sum into DEN.
__global__ void edge_softmax(const int* __restrict__ ei, int E, int T,
                             float* __restrict__ Ebuf, const float* __restrict__ EMAX,
                             float* __restrict__ DEN) {
  int e = blockIdx.x * blockDim.x + threadIdx.x;
  if (e >= T) return;
  int src, dst;
  edge_ids(ei, e, E, src, dst);
  float ee = __expf(Ebuf[e] - EMAX[dst]);
  Ebuf[e] = ee;
  atomicAdd(&DEN[dst], ee);
}

// Pass 3: out[dst] += xl[src] * (ee / denom[dst])  (atomic f32 adds -> L2)
__global__ void edge_aggregate(const int* __restrict__ ei, int E, int T,
                               const float* __restrict__ XL, const float* __restrict__ Ebuf,
                               const float* __restrict__ DEN, int C,
                               float* __restrict__ OUT) {
  int e = blockIdx.x * blockDim.x + threadIdx.x;
  if (e >= T) return;
  int src, dst;
  edge_ids(ei, e, E, src, dst);
  float alpha = Ebuf[e] / DEN[dst];
  const float4* pl = (const float4*)(XL + (long)src * C);
  float* po = OUT + (long)dst * C;
  for (int q = 0; q < C / 4; ++q) {
    float4 l = pl[q];
    atomicAdd(po + 4 * q + 0, l.x * alpha);
    atomicAdd(po + 4 * q + 1, l.y * alpha);
    atomicAdd(po + 4 * q + 2, l.z * alpha);
    atomicAdd(po + 4 * q + 3, l.w * alpha);
  }
}

__global__ void bias_relu(float* __restrict__ H, const float* __restrict__ b, long n,
                          int cmask, int do_relu) {
  long i = (long)blockIdx.x * blockDim.x + threadIdx.x;
  if (i >= n) return;
  float v = H[i] + b[(int)i & cmask];
  H[i] = do_relu ? (v > 0.0f ? v : 0.0f) : v;
}

// ---------------------------------------------------------------------------
// Launch
// ---------------------------------------------------------------------------
extern "C" void kernel_launch(void* const* d_in, const int* in_sizes, int n_in,
                              void* d_out, int out_size, void* d_ws, size_t ws_size,
                              hipStream_t stream) {
  (void)n_in; (void)out_size; (void)ws_size;
  const float* x = (const float*)d_in[0];
  const int* ei = (const int*)d_in[1];
  // per-layer params: Wl, bl, Wr, br, att, b  at base 2 (sh), 8 (mu), 14 (lv)
  const float* P[18];
  for (int i = 0; i < 18; ++i) P[i] = (const float*)d_in[2 + i];

  const int n = in_sizes[0] / 256;  // nodes
  const int E = in_sizes[1] / 2;    // edges (pre self-loop)
  const int T = E + n;              // total edges incl. self loops

  // Workspace layout (f32)
  float* XL = (float*)d_ws;               // n x 128 (reused at n x 64)
  float* XR = XL + (size_t)n * 128;       // n x 128
  float* H  = XR + (size_t)n * 128;       // n x 128 (layer-1 output)
  float* Eb = H + (size_t)n * 128;        // T
  float* EMAX = Eb + (size_t)T;           // n
  float* DEN = EMAX + (size_t)n;          // n

  float* mu_out = (float*)d_out;                 // n x 64
  float* lv_out = (float*)d_out + (size_t)n * 64;

  const int eb = 256;
  const int egrid = (T + eb - 1) / eb;
  dim3 gg((n + 63) / 64);

  // ---------------- Layer 1: sh (256 -> 128), relu ----------------
  {
    const float *Wl = P[0], *bl = P[1], *Wr = P[2], *br = P[3], *att = P[4], *b = P[5];
    gemm_bias_wmma<256, 128><<<gg, 256, 0, stream>>>(x, Wl, bl, XL, n);
    gemm_bias_wmma<256, 128><<<gg, 256, 0, stream>>>(x, Wr, br, XR, n);
    fill_f32<<<256, 256, 0, stream>>>(EMAX, n, -INFINITY);
    fill_f32<<<256, 256, 0, stream>>>(DEN, n, 0.0f);
    fill_f32<<<1024, 256, 0, stream>>>(H, (long)n * 128, 0.0f);
    edge_logits<<<egrid, eb, 0, stream>>>(ei, E, T, XL, XR, att, 128, Eb, EMAX);
    edge_softmax<<<egrid, eb, 0, stream>>>(ei, E, T, Eb, EMAX, DEN);
    edge_aggregate<<<egrid, eb, 0, stream>>>(ei, E, T, XL, Eb, DEN, 128, H);
    long tot = (long)n * 128;
    bias_relu<<<(int)((tot + 255) / 256), 256, 0, stream>>>(H, b, tot, 127, 1);
  }

  // ---------------- Layer 2: mu (128 -> 64) ----------------
  {
    const float *Wl = P[6], *bl = P[7], *Wr = P[8], *br = P[9], *att = P[10], *b = P[11];
    gemm_bias_wmma<128, 64><<<gg, 128, 0, stream>>>(H, Wl, bl, XL, n);
    gemm_bias_wmma<128, 64><<<gg, 128, 0, stream>>>(H, Wr, br, XR, n);
    fill_f32<<<256, 256, 0, stream>>>(EMAX, n, -INFINITY);
    fill_f32<<<256, 256, 0, stream>>>(DEN, n, 0.0f);
    fill_f32<<<1024, 256, 0, stream>>>(mu_out, (long)n * 64, 0.0f);
    edge_logits<<<egrid, eb, 0, stream>>>(ei, E, T, XL, XR, att, 64, Eb, EMAX);
    edge_softmax<<<egrid, eb, 0, stream>>>(ei, E, T, Eb, EMAX, DEN);
    edge_aggregate<<<egrid, eb, 0, stream>>>(ei, E, T, XL, Eb, DEN, 64, mu_out);
    long tot = (long)n * 64;
    bias_relu<<<(int)((tot + 255) / 256), 256, 0, stream>>>(mu_out, b, tot, 63, 0);
  }

  // ---------------- Layer 3: lv (128 -> 64) ----------------
  {
    const float *Wl = P[12], *bl = P[13], *Wr = P[14], *br = P[15], *att = P[16], *b = P[17];
    gemm_bias_wmma<128, 64><<<gg, 128, 0, stream>>>(H, Wl, bl, XL, n);
    gemm_bias_wmma<128, 64><<<gg, 128, 0, stream>>>(H, Wr, br, XR, n);
    fill_f32<<<256, 256, 0, stream>>>(EMAX, n, -INFINITY);
    fill_f32<<<256, 256, 0, stream>>>(DEN, n, 0.0f);
    fill_f32<<<1024, 256, 0, stream>>>(lv_out, (long)n * 64, 0.0f);
    edge_logits<<<egrid, eb, 0, stream>>>(ei, E, T, XL, XR, att, 64, Eb, EMAX);
    edge_softmax<<<egrid, eb, 0, stream>>>(ei, E, T, Eb, EMAX, DEN);
    edge_aggregate<<<egrid, eb, 0, stream>>>(ei, E, T, XL, Eb, DEN, 64, lv_out);
    long tot = (long)n * 64;
    bias_relu<<<(int)((tot + 255) / 256), 256, 0, stream>>>(lv_out, b, tot, 63, 0);
  }
}
